// CrossSpatialLearning_7842610282820
// MI455X (gfx1250) — matrix operand order repaired and checked by reading, compiled, not run on hardware
//
#include <hip/hip_runtime.h>
#include <hip/hip_bf16.h>

#define BATCH 4
#define CCH   256
#define NGRP  4
#define CPG   64
#define HH    64
#define WW    64
#define HWSZ  4096

typedef __attribute__((ext_vector_type(16))) __bf16 v16bf;
typedef __attribute__((ext_vector_type(8)))  __bf16 v8bf;
typedef __attribute__((ext_vector_type(8)))  float  v8f;
typedef int v4i __attribute__((vector_size(16)));   // matches builtin 'V4i'

// ---------------- async global->LDS copy (gfx1250 path, guarded) ----------------
#if __has_builtin(__builtin_amdgcn_global_load_async_to_lds_b128)
#define HAVE_ASYNC_LDS 1
#else
#define HAVE_ASYNC_LDS 0
#endif

__device__ __forceinline__ void cp16(__bf16* dst_lds, const __bf16* src_glob) {
#if HAVE_ASYNC_LDS
  __builtin_amdgcn_global_load_async_to_lds_b128(
      (__attribute__((address_space(1))) v4i*)src_glob,
      (__attribute__((address_space(3))) v4i*)dst_lds, 0, 0);
#else
  *(v4i*)dst_lds = *(const v4i*)src_glob;
#endif
}

__device__ __forceinline__ void async_join_barrier() {
#if HAVE_ASYNC_LDS
  asm volatile("s_wait_asynccnt 0x0" ::: "memory");
#endif
  __syncthreads();
}

// ---------------- WMMA fragment loaders ----------------
__device__ __forceinline__ v8f wmma_bf16(v16bf a, v16bf b, v8f c) {
  return __builtin_amdgcn_wmma_f32_16x16x32_bf16(false, a, false, b, (short)0, c, false, false);
}

// A fragment (16x32 bf16, MxK). LDS layout row-major [m][k], stride ld.
__device__ __forceinline__ v16bf load_A(const __bf16* M, int ld, int m_base, int k_base, int lane) {
  const int m  = m_base + (lane & 15);
  const int kh = (lane >> 4) << 4;
  const __bf16* p = M + m * ld + k_base;
  v16bf a;
#pragma unroll
  for (int j = 0; j < 8; ++j) { a[j] = p[kh + j]; a[8 + j] = p[kh + 8 + j]; }
  return a;
}

// B fragment (32x16 bf16, KxN). LDS layout [n][k] (k contiguous), stride ld.
__device__ __forceinline__ v16bf load_B(const __bf16* M, int ld, int n_base, int k_base, int lane) {
  const __bf16* p = M + (n_base + (lane & 15)) * ld + k_base + ((lane >> 4) << 4);
  v16bf b;
#pragma unroll
  for (int j = 0; j < 16; ++j) b[j] = p[j];
  return b;
}

// ---------- K0: per-(b,c) row means (xh) and column means (xw) -> xc[bc][128]
__global__ void k0_pool(const float* __restrict__ x, float* __restrict__ xc) {
  const int bc = blockIdx.x;
  const int t  = threadIdx.x;           // 0..63
  const float* xp = x + (size_t)bc * HWSZ;
  float cs = 0.f, rs = 0.f;
#pragma unroll 8
  for (int h = 0; h < HH; ++h) cs += xp[h * WW + t];
#pragma unroll 8
  for (int w = 0; w < WW; ++w) rs += xp[t * WW + w];
  xc[(size_t)bc * 128 + t]      = rs * (1.f / 64.f);   // xh (mean over W)
  xc[(size_t)bc * 128 + 64 + t] = cs * (1.f / 64.f);   // xw (mean over H)
}

// ---------- K1: y = w2@xc + b2; x2 = h2 (outer) w2s -> x2 [ch][hw] and x2t [hw][ch]
__global__ void k1_excite(const float* __restrict__ w2, const float* __restrict__ b2,
                          const float* __restrict__ xc,
                          __bf16* __restrict__ x2out, __bf16* __restrict__ x2t) {
  __shared__ alignas(16) __bf16 w2l[64 * 64];    // [o][i]
  __shared__ alignas(16) __bf16 xcs[64 * 128];   // [i][l]
  __shared__ float  yl[64 * 128];                // [o][l]
  const int bg = blockIdx.x, b = bg >> 2, g = bg & 3;
  const int t = threadIdx.x;         // 0..127
  for (int f = t; f < 4096; f += 128) w2l[f] = (__bf16)w2[g * 4096 + f];
  for (int f = t; f < 8192; f += 128) {
    int i = f >> 7, l = f & 127;
    xcs[f] = (__bf16)xc[(size_t)(b * CCH + g * CPG + i) * 128 + l];
  }
  __syncthreads();
  for (int o = 0; o < 64; ++o) {
    float s = b2[g * CPG + o];
#pragma unroll 8
    for (int i = 0; i < 64; ++i) s += (float)w2l[o * 64 + i] * (float)xcs[i * 128 + t];
    yl[o * 128 + t] = s;
  }
  __syncthreads();
  __bf16* dst  = x2out + (size_t)(b * CCH + g * CPG) * HWSZ;
  __bf16* dstt = x2t + (size_t)bg * HWSZ * 64;
  for (int c = 0; c < 64; ++c) {
    const float* yc = &yl[c * 128];
    for (int idx = t; idx < HWSZ; idx += 128) {
      int h = idx >> 6, w = idx & 63;
      __bf16 v = (__bf16)(yc[h] * yc[64 + w]);
      dst[(size_t)c * HWSZ + idx] = v;
      dstt[(size_t)idx * 64 + c]  = v;
    }
  }
}

// ---------- K2: a = w1@xg + b1 -> a_t [bg][hw][ch] (bf16), WMMA
__global__ void k2_mix(const float* __restrict__ x, const float* __restrict__ w1,
                       const float* __restrict__ b1, __bf16* __restrict__ at) {
  __shared__ alignas(16) __bf16 W1L[64 * 64];   // [o][i]
  __shared__ alignas(16) __bf16 XL[128 * 64];   // [n][i]
  const int n0 = blockIdx.x * 128;
  const int bg = blockIdx.y, b = bg >> 2, g = bg & 3;
  const int tid = threadIdx.x, lane = tid & 31, wv = tid >> 5;
  for (int f = tid; f < 4096; f += 256) W1L[f] = (__bf16)w1[g * 4096 + f];
  for (int f = tid; f < 8192; f += 256) {
    int i = f >> 7, n = f & 127;
    XL[n * 64 + i] = (__bf16)x[(size_t)(b * CCH + g * CPG + i) * HWSZ + n0 + n];
  }
  __syncthreads();
  const int m_base = (wv >> 1) * 16;
  const int n_half = (wv & 1) * 64;
  const int mrow = m_base + ((lane >> 4) << 3);      // first of 8 consecutive m
  __bf16* atp = at + (size_t)bg * HWSZ * 64;
  v16bf A0 = load_A(W1L, 64, m_base, 0, lane);
  v16bf A1 = load_A(W1L, 64, m_base, 32, lane);
  for (int nt = 0; nt < 4; ++nt) {
    const int nb = n_half + nt * 16;
    v8f acc = {};
    acc = wmma_bf16(A0, load_B(XL, 64, nb, 0, lane), acc);
    acc = wmma_bf16(A1, load_B(XL, 64, nb, 32, lane), acc);
    const int col = n0 + nb + (lane & 15);
    v8bf r;
#pragma unroll
    for (int v = 0; v < 8; ++v) r[v] = (__bf16)(acc[v] + b1[g * CPG + mrow + v]);
    *(v8bf*)(atp + (size_t)col * 64 + mrow) = r;     // transposed, 16B store
  }
}

// ---------- K3: grouped 3x3 conv as 9 accumulated tap GEMMs -> c_t [bg][hw][ch]
__global__ void k3_conv(const float* __restrict__ x, const float* __restrict__ w3,
                        const float* __restrict__ b3, __bf16* __restrict__ ct) {
  __shared__ alignas(16) __bf16 W3t[64 * 64];      // one tap: [oc][i]
  __shared__ alignas(16) __bf16 XH[64 * 4 * 66];   // halo: [i][r][wc]
  const int h0 = blockIdx.x * 2;
  const int bg = blockIdx.y, b = bg >> 2, g = bg & 3;
  const int tid = threadIdx.x, lane = tid & 31, wv = tid >> 5;
  for (int f = tid; f < 64 * 4 * 66; f += 256) {
    int i = f / 264, rem = f - i * 264;
    int r = rem / 66, wc = rem - r * 66;
    int h = h0 - 1 + r, wg = wc - 1;
    float v = 0.f;
    if ((unsigned)h < 64u && (unsigned)wg < 64u)
      v = x[(size_t)(b * CCH + g * CPG + i) * HWSZ + h * 64 + wg];
    XH[f] = (__bf16)v;
  }
  const int m_base = (wv >> 1) * 16;
  const int hl = wv & 1;
  v8f acc[4] = {};
  for (int tap = 0; tap < 9; ++tap) {
    __syncthreads();
    for (int f = tid; f < 4096; f += 256) {
      int oc = f >> 6, i = f & 63;
      W3t[f] = (__bf16)w3[(size_t)((g * CPG + oc) * 64 + i) * 9 + tap];
    }
    __syncthreads();
    const int ky = tap / 3, kx = tap - ky * 3;
    const int r = hl + ky;
    v16bf A0 = load_A(W3t, 64, m_base, 0, lane);
    v16bf A1 = load_A(W3t, 64, m_base, 32, lane);
#pragma unroll
    for (int nt = 0; nt < 4; ++nt) {
      const int col = r * 66 + nt * 16 + (lane & 15) + kx;
      const int kh = (lane >> 4) << 4;
      v16bf B0, B1;
#pragma unroll
      for (int j = 0; j < 16; ++j) {
        B0[j] = XH[(kh + j) * 264 + col];
        B1[j] = XH[(32 + kh + j) * 264 + col];
      }
      acc[nt] = wmma_bf16(A0, B0, acc[nt]);
      acc[nt] = wmma_bf16(A1, B1, acc[nt]);
    }
  }
  const int hw_row = (h0 + hl) * 64;
  const int mrow = m_base + ((lane >> 4) << 3);
  __bf16* ctp = ct + (size_t)bg * HWSZ * 64;
#pragma unroll
  for (int nt = 0; nt < 4; ++nt) {
    const int hw = hw_row + nt * 16 + (lane & 15);
    v8bf r;
#pragma unroll
    for (int v = 0; v < 8; ++v) r[v] = (__bf16)(acc[nt][v] + b3[g * CPG + mrow + v]);
    *(v8bf*)(ctp + (size_t)hw * 64 + mrow) = r;
  }
}

// ---------- K4: fused attention: attd = sigmoid(a^T c) @ x2^T -> attd_t [bg][hw][ch]
__global__ void k4_attn(const __bf16* __restrict__ at, const __bf16* __restrict__ ct,
                        const __bf16* __restrict__ x2_, __bf16* __restrict__ attdt) {
  __shared__ alignas(16) __bf16 QT[64 * 64];   // [m][i]
  __shared__ alignas(16) __bf16 CF[64 * 64];   // [n][i]
  __shared__ alignas(16) __bf16 XT[64 * 64];   // [cc][n]
  __shared__ alignas(16) __bf16 PT[64 * 64];   // [m][n]
  const int m0 = blockIdx.x * 64;
  const int bg = blockIdx.y, b = bg >> 2, g = bg & 3;
  const int tid = threadIdx.x, lane = tid & 31, wv = tid >> 5;   // 4 waves
  const __bf16* atp = at + (size_t)bg * HWSZ * 64;
  const __bf16* ctp = ct + (size_t)bg * HWSZ * 64;
  const __bf16* xp  = x2_ + (size_t)(b * CCH + g * CPG) * HWSZ;
  // Q tile: contiguous 8KB block copy (async)
  for (int f = tid; f < 512; f += 128) cp16(QT + f * 8, atp + (size_t)m0 * 64 + f * 8);
  async_join_barrier();
  const int m_base = wv * 16;
  v16bf QA0 = load_A(QT, 64, m_base, 0, lane);
  v16bf QA1 = load_A(QT, 64, m_base, 32, lane);
  v8f acc[4] = {};
  for (int n0 = 0; n0 < HWSZ; n0 += 64) {
    __syncthreads();                       // previous-iteration LDS reads done
    for (int f = tid; f < 512; f += 128)   // K tile: contiguous 8KB
      cp16(CF + f * 8, ctp + (size_t)n0 * 64 + f * 8);
    for (int f = tid; f < 512; f += 128) { // V tile: 64 rows x 128B
      int cc = f >> 3, q = f & 7;
      cp16(XT + f * 8, xp + (size_t)cc * HWSZ + n0 + q * 8);
    }
    async_join_barrier();
#pragma unroll
    for (int nt = 0; nt < 4; ++nt) {
      v8f s = {};
      s = wmma_bf16(QA0, load_B(CF, 64, nt * 16, 0, lane), s);
      s = wmma_bf16(QA1, load_B(CF, 64, nt * 16, 32, lane), s);
      const int ncol = nt * 16 + (lane & 15);
#pragma unroll
      for (int v = 0; v < 8; ++v) {
        int m = m_base + v + ((lane >> 4) << 3);
        PT[m * 64 + ncol] = (__bf16)(1.f / (1.f + __expf(-s[v])));
      }
    }
    __syncthreads();
    v16bf PA0 = load_A(PT, 64, m_base, 0, lane);
    v16bf PA1 = load_A(PT, 64, m_base, 32, lane);
#pragma unroll
    for (int ct_ = 0; ct_ < 4; ++ct_) {
      acc[ct_] = wmma_bf16(PA0, load_B(XT, 64, ct_ * 16, 0, lane), acc[ct_]);
      acc[ct_] = wmma_bf16(PA1, load_B(XT, 64, ct_ * 16, 32, lane), acc[ct_]);
    }
  }
  __bf16* op = attdt + (size_t)bg * HWSZ * 64;
#pragma unroll
  for (int ct_ = 0; ct_ < 4; ++ct_) {
    const int cc = ct_ * 16 + (lane & 15);
#pragma unroll
    for (int v = 0; v < 8; ++v) {
      int m = m_base + v + ((lane >> 4) << 3);
      op[(size_t)(m0 + m) * 64 + cc] = (__bf16)acc[ct_][v];
    }
  }
}

// ---------- K5: out = wf @ concat_g[a|x2|attd] + bf (per b: 256x768 @ 768x4096)
__global__ void k5_proj(const float* __restrict__ wf, const float* __restrict__ bfb,
                        const __bf16* __restrict__ at, const __bf16* __restrict__ x2t,
                        const __bf16* __restrict__ attdt, float* __restrict__ out) {
  __shared__ alignas(16) __bf16 WFL[64 * 64];   // [m][k]
  __shared__ alignas(16) __bf16 UT[128 * 64];   // [n][k]
  const int n0 = blockIdx.x * 128;
  const int m0 = blockIdx.y * 64;
  const int b  = blockIdx.z;
  const int tid = threadIdx.x, lane = tid & 31, wv = tid >> 5;
  const int m_base = (wv >> 1) * 16;
  const int n_half = (wv & 1) * 64;
  v8f acc[4] = {};
  for (int k0 = 0; k0 < 768; k0 += 64) {
    __syncthreads();
    for (int f = tid; f < 4096; f += 256) {
      int m = f >> 6, k = f & 63;
      WFL[f] = (__bf16)wf[(size_t)(m0 + m) * 768 + k0 + k];
    }
    // concat axis=2: channel block ub -> group ub/3, source ub%3 in {a,x2,attd}
    const int ub = k0 >> 6;
    const int grp = ub / 3, which = ub - grp * 3;
    const __bf16* src = (which == 0) ? at : (which == 1) ? x2t : attdt;
    src += (size_t)(b * NGRP + grp) * HWSZ * 64 + (size_t)n0 * 64;
    for (int f = tid; f < 1024; f += 256)   // 16KB contiguous block copy (async)
      cp16(UT + f * 8, src + f * 8);
    async_join_barrier();
    v16bf A0 = load_A(WFL, 64, m_base, 0, lane);
    v16bf A1 = load_A(WFL, 64, m_base, 32, lane);
#pragma unroll
    for (int nt = 0; nt < 4; ++nt) {
      const int nb = n_half + nt * 16;
      acc[nt] = wmma_bf16(A0, load_B(UT, 64, nb, 0, lane), acc[nt]);
      acc[nt] = wmma_bf16(A1, load_B(UT, 64, nb, 32, lane), acc[nt]);
    }
  }
#pragma unroll
  for (int nt = 0; nt < 4; ++nt) {
    const int col = n0 + n_half + nt * 16 + (lane & 15);
#pragma unroll
    for (int v = 0; v < 8; ++v) {
      int m = m0 + m_base + v + ((lane >> 4) << 3);
      out[(size_t)(b * CCH + m) * HWSZ + col] = acc[nt][v] + bfb[m];
    }
  }
}

extern "C" void kernel_launch(void* const* d_in, const int* in_sizes, int n_in,
                              void* d_out, int out_size, void* d_ws, size_t ws_size,
                              hipStream_t stream) {
  (void)in_sizes; (void)n_in; (void)out_size; (void)ws_size;
  const float* x   = (const float*)d_in[0];
  const float* w1  = (const float*)d_in[1];
  const float* b1  = (const float*)d_in[2];
  const float* w2  = (const float*)d_in[3];
  const float* b2  = (const float*)d_in[4];
  const float* w3  = (const float*)d_in[5];
  const float* b3  = (const float*)d_in[6];
  const float* wf  = (const float*)d_in[7];
  const float* bfb = (const float*)d_in[8];
  float* out = (float*)d_out;

  const size_t NBC = (size_t)BATCH * CCH * HWSZ;   // 4,194,304 elements
  char* ws = (char*)d_ws;
  __bf16* ws_at    = (__bf16*)(ws);                // a   transposed [bg][hw][ch]
  __bf16* ws_x2    = (__bf16*)(ws + NBC * 2);      // x2  [b][ch][hw]
  __bf16* ws_x2t   = (__bf16*)(ws + NBC * 4);      // x2  transposed [bg][hw][ch]
  __bf16* ws_ct    = (__bf16*)(ws + NBC * 6);      // c   transposed [bg][hw][ch]
  __bf16* ws_attdt = (__bf16*)(ws + NBC * 8);      // attd transposed [bg][hw][ch]
  float*  ws_xc    = (float*) (ws + NBC * 10);     // B*C*128 f32

  k0_pool  <<<dim3(BATCH * CCH),                 dim3(64),  0, stream>>>(x, ws_xc);
  k1_excite<<<dim3(BATCH * NGRP),                dim3(128), 0, stream>>>(w2, b2, ws_xc, ws_x2, ws_x2t);
  k2_mix   <<<dim3(HWSZ / 128, BATCH * NGRP),    dim3(256), 0, stream>>>(x, w1, b1, ws_at);
  k3_conv  <<<dim3(HH / 2, BATCH * NGRP),        dim3(256), 0, stream>>>(x, w3, b3, ws_ct);
  k4_attn  <<<dim3(HWSZ / 64, BATCH * NGRP),     dim3(128), 0, stream>>>(ws_at, ws_ct, ws_x2, ws_attdt);
  k5_proj  <<<dim3(HWSZ / 128, CCH / 64, BATCH), dim3(256), 0, stream>>>(wf, bfb, ws_at, ws_x2t, ws_attdt, out);
}
